// Decoder__90237262889676
// MI455X (gfx1250) — compile-verified, matrix operand order checked by reading, not run on hardware
//
#include <hip/hip_runtime.h>

// ---------------------------------------------------------------------------
// CDNA5 (gfx1250) WMMA implementation of the sparse-conv decoder.
// GEMM work (3x3x3 conv, NiN skip, 2x2x2 deconv) -> v_wmma_f32_16x16x32_bf16.
// Conv kernel: 4 waves/block, each wave = 16 voxels x 64 out-channels
// (4 accumulators). Packed B fragments are double-buffered through LDS with
// global_load_async_to_lds_b128 (ASYNCcnt) and read back with ds_load_b128.
// ---------------------------------------------------------------------------

typedef __bf16 v16bf __attribute__((ext_vector_type(16)));
typedef __bf16 v8bf  __attribute__((ext_vector_type(8)));
typedef float  v8f   __attribute__((ext_vector_type(8)));

#define EPS 1e-4f

// mask0 is [B,16,16,16]; msh = log2(D/16) for the current resolution D.
__device__ __forceinline__ float mask_at(const int* __restrict__ m0,
                                         int b, int d, int h, int w, int msh) {
  int idx = ((((b << 4) | (d >> msh)) << 4) | (h >> msh));
  idx = ((idx << 4) | (w >> msh));
  return m0[idx] ? 1.0f : 0.0f;
}

// ---------------------------------------------------------------------------
// Weight repack: f32 [tap][k][n] -> bf16 WMMA B fragments
// [tap][kb][nb][lane][16 halves], matching the 16-bit 32x16 B layout:
//   lane l (l<16): col n=l,     K = kb*32 + {0..7, 16..23}
//   lane l (>=16): col n=l-16,  K = kb*32 + {8..15, 24..31}
// ---------------------------------------------------------------------------
__global__ void k_pack(const float* __restrict__ W, __bf16* __restrict__ P,
                       int nelem, int KB, int NB, int cout) {
  int i = blockIdx.x * blockDim.x + threadIdx.x;
  if (i >= nelem) return;
  int f = i >> 9;
  int r = i & 511;
  int l = r >> 4;
  int j = r & 15;
  int fm  = f % (KB * NB);
  int tap = f / (KB * NB);
  int kb  = fm / NB;
  int nb  = fm % NB;
  int n   = nb * 16 + (l & 15);
  int kc  = l >> 4;
  int k   = kb * 32 + kc * 8 + (j < 8 ? j : j + 8);
  int cin = KB * 32;
  P[i] = (__bf16)W[((long)tap * cin + k) * cout + n];
}

// ---------------------------------------------------------------------------
// Implicit-GEMM conv / NiN GEMM.  ntaps==27 -> 3x3x3 SAME conv; ntaps==1 -> GEMM.
// grid.x = nvox/64, block = 128 (4 waves; wave w owns voxel tile blk*4+w and
// computes all 64 out channels -> 4 WMMA accumulators per wave).
// B fragments: async global->LDS double buffer, one tap ahead.
// ---------------------------------------------------------------------------
__global__ void __launch_bounds__(128)
k_conv_wmma(const __bf16* __restrict__ act, const __bf16* __restrict__ wp,
            float* __restrict__ out, int ntaps, int KB, int cin, int log2D) {
  __shared__ __align__(16) char shB[2][16384];  // per-tap B frags (<=16KB), x2

  const int tid  = threadIdx.x;
  const int wave = tid >> 5;
  const int lane = tid & 31;
  const int row  = lane & 15;
  const int kc   = lane >> 4;
  const int D    = 1 << log2D;
  const int Dm   = D - 1;

  const long vbase = ((long)blockIdx.x * 4 + wave) * 16;
  const long v     = vbase + row;
  const int w = (int)(v) & Dm;
  const int h = (int)(v >> log2D) & Dm;
  const int d = (int)(v >> (2 * log2D)) & Dm;
  const int b = (int)(v >> (3 * log2D));

  const unsigned ldsbase   = (unsigned)(unsigned long long)&shB[0][0];
  const unsigned long long wpu = (unsigned long long)wp;
  const int tapBytes = KB << 12;        // KB * 4KB per tap
  const int rounds   = tapBytes >> 11;  // 128 thr x 16B = 2KB per round

  // Kick async copy of one tap's packed B fragments into LDS buffer `buf`.
  auto kick = [&](int tap, int buf) {
    const unsigned gbase = (unsigned)(tap * tapBytes) + ((unsigned)tid << 4);
    const unsigned lbase = ldsbase + (unsigned)(buf << 14) + ((unsigned)tid << 4);
    for (int r = 0; r < rounds; ++r) {
      const unsigned lo = lbase + ((unsigned)r << 11);
      const unsigned go = gbase + ((unsigned)r << 11);
      asm volatile("global_load_async_to_lds_b128 %0, %1, %2"
                   :: "v"(lo), "v"(go), "s"(wpu) : "memory");
    }
  };

  kick(0, 0);

  v8f acc[4];
#pragma unroll
  for (int nb = 0; nb < 4; ++nb) acc[nb] = (v8f){};

  for (int t = 0; t < ntaps; ++t) {
    const int buf = t & 1;
    asm volatile("s_wait_asynccnt 0x0" ::: "memory");  // this buf's copies done
    __syncthreads();                                   // visible to all waves
    if (t + 1 < ntaps) kick(t + 1, buf ^ 1);

    int dz = 0, dy = 0, dx = 0;
    if (ntaps == 27) { dz = t / 9 - 1; dy = (t / 3) % 3 - 1; dx = t % 3 - 1; }
    const int d2 = d + dz, h2 = h + dy, w2 = w + dx;
    const bool ok = (unsigned)d2 < (unsigned)D &&
                    (unsigned)h2 < (unsigned)D &&
                    (unsigned)w2 < (unsigned)D;
    const __bf16* ap = act + ((((long)b * D + d2) * D + h2) * D + w2) * cin;

    for (int kb = 0; kb < KB; ++kb) {
      v16bf afrag = {};
      if (ok) {
        const int kbase = kb * 32 + kc * 8;
        v8bf lo = *(const v8bf*)(ap + kbase);
        v8bf hi = *(const v8bf*)(ap + kbase + 16);
#pragma unroll
        for (int j = 0; j < 8; ++j) { afrag[j] = lo[j]; afrag[8 + j] = hi[j]; }
      }
#pragma unroll
      for (int nb = 0; nb < 4; ++nb) {
        // frag = [kb][nb], 1KB each; lane stride 32B -> ds_load_b128 pair
        v16bf bfrag = *(const v16bf*)(shB[buf] + ((kb * 4 + nb) << 10) +
                                      (lane << 5));
        acc[nb] = __builtin_amdgcn_wmma_f32_16x16x32_bf16(
            false, afrag, false, bfrag, (short)0, acc[nb], false, false);
      }
    }
  }

  // D layout: lane<16 -> N=lane, rows 0..7; lane>=16 -> N=lane-16, rows 8..15
  const int col = lane & 15;
  float* op = out + (vbase + kc * 8) * 64 + col;
#pragma unroll
  for (int nb = 0; nb < 4; ++nb)
#pragma unroll
    for (int j = 0; j < 8; ++j) op[j * 64 + nb * 16] = acc[nb][j];
}

// ---------------------------------------------------------------------------
// 2x2x2 stride-2 deconv: 8 independent GEMMs (K=64, N=64), A reused across taps.
// grid.x = nvox_in/16, block = 128.
// ---------------------------------------------------------------------------
__global__ void __launch_bounds__(128)
k_deconv_wmma(const __bf16* __restrict__ act, const __bf16* __restrict__ wp,
              float* __restrict__ out, int log2D, int ostride, int choff) {
  const int wave = threadIdx.x >> 5;
  const int lane = threadIdx.x & 31;
  const int row  = lane & 15;
  const int kc   = lane >> 4;
  const int D    = 1 << log2D;
  const int Dm   = D - 1;
  const int D2   = D << 1;

  const long vbase = (long)blockIdx.x * 16;
  const long v     = vbase + row;
  const __bf16* ap = act + v * 64;

  v8f acc[8];
#pragma unroll
  for (int t = 0; t < 8; ++t) acc[t] = (v8f){};

#pragma unroll
  for (int kb = 0; kb < 2; ++kb) {
    const int kbase = kb * 32 + kc * 8;
    v8bf lo = *(const v8bf*)(ap + kbase);
    v8bf hi = *(const v8bf*)(ap + kbase + 16);
    v16bf afrag;
#pragma unroll
    for (int j = 0; j < 8; ++j) { afrag[j] = lo[j]; afrag[8 + j] = hi[j]; }
#pragma unroll
    for (int t = 0; t < 8; ++t) {
      v16bf bfrag = *(const v16bf*)(wp + ((long)(t * 2 + kb) * 4 + wave) * 512 +
                                    lane * 16);
      acc[t] = __builtin_amdgcn_wmma_f32_16x16x32_bf16(
          false, afrag, false, bfrag, (short)0, acc[t], false, false);
    }
  }

  const int col = lane & 15;
#pragma unroll
  for (int j = 0; j < 8; ++j) {
    const long vj = vbase + j + kc * 8;
    const int wj = (int)(vj) & Dm;
    const int hj = (int)(vj >> log2D) & Dm;
    const int dj = (int)(vj >> (2 * log2D)) & Dm;
    const int bj = (int)(vj >> (3 * log2D));
#pragma unroll
    for (int t = 0; t < 8; ++t) {
      const int i0 = t >> 2, j0 = (t >> 1) & 1, k0 = t & 1;
      const long ov = ((((long)bj * D2 + 2 * dj + i0) * D2 + 2 * hj + j0) * D2) +
                      2 * wj + k0;
      out[ov * (long)ostride + choff + wave * 16 + col] = acc[t][j];
    }
  }
}

// ---------------------------------------------------------------------------
// Elementwise kernels
// ---------------------------------------------------------------------------
__global__ void k_mask_x0(const float* __restrict__ x0, const int* __restrict__ m0,
                          float* __restrict__ ret0, float* __restrict__ xf,
                          __bf16* __restrict__ xb, int total) {
  int i = blockIdx.x * blockDim.x + threadIdx.x;
  if (i >= total) return;
  int v = i >> 7;  // C=128
  float m = m0[v] ? 1.0f : 0.0f;
  float val = x0[i] * m;
  ret0[i] = val;
  xf[i] = val;
  xb[i] = (__bf16)val;
}

__global__ void k_bnrelu(const float* __restrict__ x, const float* __restrict__ gg,
                         const float* __restrict__ bb, const float* __restrict__ mn,
                         const float* __restrict__ va, const int* __restrict__ m0,
                         __bf16* __restrict__ outb, int total, int log2C,
                         int log2D, int msh) {
  int i = blockIdx.x * blockDim.x + threadIdx.x;
  if (i >= total) return;
  int c = i & ((1 << log2C) - 1);
  int v = i >> log2C;
  int Dm = (1 << log2D) - 1;
  int w = v & Dm;
  int h = (v >> log2D) & Dm;
  int d = (v >> (2 * log2D)) & Dm;
  int b = v >> (3 * log2D);
  float m = mask_at(m0, b, d, h, w, msh);
  float s = gg[c] * rsqrtf(va[c] + EPS);
  float val = (x[i] - mn[c]) * s + bb[c];
  outb[i] = (__bf16)(fmaxf(val, 0.0f) * m);
}

__global__ void k_skipadd(const float* __restrict__ skip, const float* __restrict__ cv,
                          const int* __restrict__ m0, float* __restrict__ xf,
                          __bf16* __restrict__ xb, int total, int log2D, int msh) {
  int i = blockIdx.x * blockDim.x + threadIdx.x;
  if (i >= total) return;
  int v = i >> 6;  // C=64
  int Dm = (1 << log2D) - 1;
  int w = v & Dm;
  int h = (v >> log2D) & Dm;
  int d = (v >> (2 * log2D)) & Dm;
  int b = v >> (3 * log2D);
  float m = mask_at(m0, b, d, h, w, msh);
  float val = skip[i] + cv[i] * m;
  xf[i] = val;
  xb[i] = (__bf16)val;
}

// Finish stage-0 output: deconv already wrote channels 0..63 of xf (stride 128);
// write x1*m1 into channels 64..127, mirror to bf16, and emit ret1.
__global__ void k_concat(const float* __restrict__ x1, const int* __restrict__ m0,
                         float* __restrict__ xf, __bf16* __restrict__ xb,
                         float* __restrict__ ret1, int total) {
  int i = blockIdx.x * blockDim.x + threadIdx.x;
  if (i >= total) return;
  int c = i & 127;
  int v = i >> 7;
  int w = v & 31;
  int h = (v >> 5) & 31;
  int d = (v >> 10) & 31;
  int b = v >> 15;
  float val;
  if (c < 64) {
    val = xf[i];
  } else {
    float m = mask_at(m0, b, d, h, w, 1);
    val = x1[(long)v * 64 + (c - 64)] * m;
    xf[i] = val;
  }
  xb[i] = (__bf16)val;
  ret1[i] = val;
}

// ---------------------------------------------------------------------------
// Host side
// ---------------------------------------------------------------------------
extern "C" void kernel_launch(void* const* d_in, const int* in_sizes, int n_in,
                              void* d_out, int out_size, void* d_ws, size_t ws_size,
                              hipStream_t stream) {
  (void)in_sizes; (void)n_in; (void)out_size; (void)ws_size;

  const float* x0 = (const float*)d_in[0];
  const float* x1 = (const float*)d_in[1];
  const int*   m0 = (const int*)d_in[2];

  // Workspace layout (bytes)
  char* ws = (char*)d_ws;
  __bf16* WP  = (__bf16*)(ws);                     //  2.5 MB packed weights
  float*  XF  = (float*)(ws + 2621440);            // 67 MB f32 master act
  __bf16* XB  = (__bf16*)(ws + 2621440 + 67108864);             // bf16 act
  __bf16* HB  = (__bf16*)(ws + 2621440 + 67108864 + 33554432);  // bn-relu out
  float*  C1  = (float*)(ws + 2621440 + 67108864 + 2 * 33554432);  // conv out
  float*  SKP = (float*)(ws + 2621440 + 67108864 + 2 * 33554432 + 33554432);

  float* ret0 = (float*)d_out;
  float* ret1 = ret0 + 2097152;    // [4,32,32,32,128]
  float* ret2 = ret1 + 16777216;   // [4,64,64,64,64]

  // Packed-weight offsets (elements)
  const long L0C1 = 0,       L0C2 = 221184,  L0NIN = 331776;
  const long L0R1C1 = 339968, L0R1C2 = 450560, L0R1NIN = 561152, L0DEC = 565248;
  const long L1C1 = 598016,  L1C2 = 819200,  L1NIN = 929792;
  const long L1R1C1 = 937984, L1R1C2 = 1048576, L1R1NIN = 1159168, L1DEC = 1163264;

  auto pack = [&](int idx, long off, int taps, int KB) {
    int nelem = taps * KB * 32 * 64;
    k_pack<<<(nelem + 255) / 256, 256, 0, stream>>>(
        (const float*)d_in[idx], WP + off, nelem, KB, 4, 64);
  };
  pack(12, L0C1, 27, 4);  pack(13, L0C2, 27, 2);  pack(14, L0NIN, 1, 4);
  pack(23, L0R1C1, 27, 2); pack(24, L0R1C2, 27, 2); pack(25, L0R1NIN, 1, 2);
  pack(3,  L0DEC, 8, 2);
  pack(35, L1C1, 27, 4);  pack(36, L1C2, 27, 2);  pack(37, L1NIN, 1, 4);
  pack(46, L1R1C1, 27, 2); pack(47, L1R1C2, 27, 2); pack(48, L1R1NIN, 1, 2);
  pack(26, L1DEC, 8, 2);

  auto bn = [&](const float* x, int gi, long total, int log2C, int log2D, int msh) {
    k_bnrelu<<<(int)((total + 255) / 256), 256, 0, stream>>>(
        x, (const float*)d_in[gi], (const float*)d_in[gi + 1],
        (const float*)d_in[gi + 2], (const float*)d_in[gi + 3], m0, HB,
        (int)total, log2C, log2D, msh);
  };
  auto conv = [&](const __bf16* act, long woff, float* out, int ntaps, int KB,
                  int log2D, long nvox) {
    k_conv_wmma<<<(int)(nvox / 64), 128, 0, stream>>>(
        act, WP + woff, out, ntaps, KB, KB * 32, log2D);
  };
  auto rep = [&](int bn1i, int bn2i, long c1off, long c2off, long ninoff,
                 int cin, int log2D, int msh, long nvox) {
    const int KBin = cin / 32;
    conv(XB, ninoff, SKP, 1, KBin, log2D, nvox);                // NiN skip
    bn(XF, bn1i, nvox * cin, cin == 128 ? 7 : 6, log2D, msh);   // bn1+relu+mask
    conv(HB, c1off, C1, 27, KBin, log2D, nvox);                 // conv1
    bn(C1, bn2i, nvox * 64, 6, log2D, msh);                     // bn2+relu+mask
    conv(HB, c2off, C1, 27, 2, log2D, nvox);                    // conv2
    k_skipadd<<<(int)((nvox * 64 + 255) / 256), 256, 0, stream>>>(
        SKP, C1, m0, XF, XB, (int)(nvox * 64), log2D, msh);
  };

  // ---- stage 0 @ 16^3 (16384 voxels) ----
  k_mask_x0<<<2097152 / 256, 256, 0, stream>>>(x0, m0, ret0, XF, XB, 2097152);
  rep(4, 8, L0C1, L0C2, L0NIN, 128, 4, 0, 16384);
  rep(15, 19, L0R1C1, L0R1C2, L0R1NIN, 64, 4, 0, 16384);
  k_deconv_wmma<<<16384 / 16, 128, 0, stream>>>(XB, WP + L0DEC, XF, 4, 128, 0);
  k_concat<<<16777216 / 256, 256, 0, stream>>>(x1, m0, XF, XB, ret1, 16777216);

  // ---- stage 1 @ 32^3 (131072 voxels) ----
  rep(27, 31, L1C1, L1C2, L1NIN, 128, 5, 1, 131072);
  rep(38, 42, L1R1C1, L1R1C2, L1R1NIN, 64, 5, 1, 131072);
  k_deconv_wmma<<<131072 / 16, 128, 0, stream>>>(XB, WP + L1DEC, ret2, 5, 64, 0);
}
